// MultiHeadSelfAttention_48850958024962
// MI455X (gfx1250) — matrix operand order repaired and checked by reading, not compiled, not run on hardware
//
#include <hip/hip_runtime.h>
#include <math.h>

// ---------------------------------------------------------------------------
// MI455X (gfx1250) multi-head self-attention, wave32 + V_WMMA_F32_16X16X32_F16.
//
// Roofline: attn output alone is 1.07 GB (must be materialized) -> ~46us at
// 23.3 TB/s.  Total matmul work is only ~77 GFLOP, far under the f16 WMMA
// ceiling, so we (a) fuse softmax + attn-write + attn@V in one pass so attn
// is written exactly once and never re-read from HBM, (b) recompute QK^T for
// the normalization pass instead of buffering scores (K tile per head is
// 128 KB -> L2 resident), (c) run everything through 16x16x32 f16 WMMA.
//
// Workspace layout (f16 elements), total ~128.3 MB:
//   Q   [32][8][1024][64]   @ 0
//   K   [32][8][1024][64]   @ 16777216
//   Vt  [32][8][64][1024]   @ 33554432   (V stored transposed per head)
//   CTX [32][1024][512]     @ 50331648
//   WT  [3][512][64]        @ 67108864   (wq/wk/wv transposed, f16)
//   DT  [64][512]           @ 67207168   (dense_w transposed, f16)
// ---------------------------------------------------------------------------

typedef __attribute__((ext_vector_type(16))) _Float16 v16h;
typedef __attribute__((ext_vector_type(8)))  _Float16 v8h;
typedef __attribute__((ext_vector_type(8)))  float    v8f;

union U16H { v16h v; v8h h[2]; _Float16 e[16]; };
union U8F  { v8f  v; float e[8]; };

#define SEQ     1024
#define DEPTH   64
#define NHEAD   8
#define DMODEL  512
#define NBATCH  32
#define NROWS   (NBATCH * SEQ)          // 32768
#define OUT_ELEMS   (NROWS * DEPTH)     // 2097152 floats, then attn follows
#define Q_OFF   0
#define K_OFF   16777216
#define VT_OFF  33554432
#define CTX_OFF 50331648
#define WT_OFF  67108864
#define DT_OFF  67207168

__device__ __forceinline__ v8f wmma16(v16h a, v16h b, v8f c) {
  // D = A(16x32 f16) x B(32x16 f16) + C(16x16 f32)
  return __builtin_amdgcn_wmma_f32_16x16x32_f16(false, a, false, b,
                                                (short)0, c, false, false);
}
__device__ __forceinline__ float redmax16(float x) {
  x = fmaxf(x, __shfl_xor(x, 1, 32));
  x = fmaxf(x, __shfl_xor(x, 2, 32));
  x = fmaxf(x, __shfl_xor(x, 4, 32));
  x = fmaxf(x, __shfl_xor(x, 8, 32));
  return x;
}
__device__ __forceinline__ float redsum16(float x) {
  x += __shfl_xor(x, 1, 32);
  x += __shfl_xor(x, 2, 32);
  x += __shfl_xor(x, 4, 32);
  x += __shfl_xor(x, 8, 32);
  return x;
}
// A-matrix (16x32 f16) element i -> K index, contiguous order (k = 16*hi + i)
__device__ __forceinline__ int a_kidx(int i, int hi) {
  return hi * 16 + i;
}

// ---------------------------------------------------------------------------
// Kernel 0: transpose + f32->f16 convert the weight matrices so every WMMA
// B-operand load in later kernels is a contiguous 32-byte per-lane vector.
// WT[sel][n][k] = w_sel[k][n]   (sel: 0=q,1=k,2=v; n<512, k<64)
// DT[n][k]      = dense_w[k][n] (n<64, k<512)
// ---------------------------------------------------------------------------
__global__ __launch_bounds__(256) void k_prep_w(
    const float* __restrict__ wq, const float* __restrict__ wk,
    const float* __restrict__ wv, const float* __restrict__ dw,
    _Float16* __restrict__ WT, _Float16* __restrict__ DT) {
  int t = blockIdx.x * 256 + threadIdx.x;
  if (t < 3 * DMODEL * DEPTH) {
    int sel = t >> 15, rem = t & 32767;
    int n = rem >> 6, k = rem & 63;
    const float* w = (sel == 0) ? wq : (sel == 1) ? wk : wv;
    WT[t] = (_Float16)w[k * DMODEL + n];
  } else {
    int rem = t - 3 * DMODEL * DEPTH;
    if (rem < DEPTH * DMODEL) {
      int n = rem >> 9, k = rem & 511;
      DT[rem] = (_Float16)dw[k * DEPTH + n];
    }
  }
}

// ---------------------------------------------------------------------------
// Kernel 1: QKV projection.  grid = (2048 row-tiles, 8 heads, 3 sel).
// One wave: 16 rows of (x + pos) x W_sel -> one head's 64 output columns.
// ---------------------------------------------------------------------------
__global__ __launch_bounds__(32) void k_qkv(
    const float* __restrict__ x, const float* __restrict__ pos,
    const float* __restrict__ bq, const float* __restrict__ bk,
    const float* __restrict__ bv, const _Float16* __restrict__ WT,
    _Float16* __restrict__ Qp, _Float16* __restrict__ Kp,
    _Float16* __restrict__ Vtp) {
  const int lane = threadIdx.x, n = lane & 15, hi = lane >> 4;
  const int rt = blockIdx.x, h = blockIdx.y, sel = blockIdx.z;
  const _Float16* W = WT + sel * (DMODEL * DEPTH);
  const float* bias = (sel == 0) ? bq : (sel == 1) ? bk : bv;

  // A tiles: rows of (x + pos_emb), f32 -> f16, contiguous-k A layout, K=64 in 2 chunks
  const int mg = rt * 16 + (lane & 15);
  const int ss0 = mg & (SEQ - 1);
  const float* xr = x + (size_t)mg * DEPTH;
  const float* pr = pos + (size_t)ss0 * DEPTH;
  U16H a[2];
#pragma unroll
  for (int kk = 0; kk < 2; ++kk)
#pragma unroll
    for (int i = 0; i < 16; ++i) {
      int k = kk * 32 + a_kidx(i, hi);
      a[kk].e[i] = (_Float16)(xr[k] + pr[k]);
    }

  v8f acc[4] = {};
#pragma unroll
  for (int kk = 0; kk < 2; ++kk)
#pragma unroll
    for (int nt = 0; nt < 4; ++nt) {
      int col = h * 64 + nt * 16 + n;
      v16h bm = *(const v16h*)(W + (size_t)col * DEPTH + kk * 32 + hi * 16);
      acc[nt] = wmma16(a[kk].v, bm, acc[nt]);
    }

#pragma unroll
  for (int nt = 0; nt < 4; ++nt) {
    int col = h * 64 + nt * 16 + n, d = nt * 16 + n;
    float bvl = bias[col];
    U8F cu; cu.v = acc[nt];
#pragma unroll
    for (int r = 0; r < 8; ++r) {
      int mg2 = rt * 16 + hi * 8 + r;
      int bb = mg2 >> 10, ss = mg2 & (SEQ - 1);
      _Float16 hv = (_Float16)(cu.e[r] + bvl);
      if (sel == 0)
        Qp[(((size_t)bb * NHEAD + h) * SEQ + ss) * DEPTH + d] = hv;
      else if (sel == 1)
        Kp[(((size_t)bb * NHEAD + h) * SEQ + ss) * DEPTH + d] = hv;
      else
        Vtp[(((size_t)bb * NHEAD + h) * DEPTH + d) * SEQ + ss] = hv;
    }
  }
}

// ---------------------------------------------------------------------------
// Kernel 2: fused attention.  grid = (64 q-tiles, 8 heads, 32 batch), 1 wave.
// Pass 1: online softmax stats over 64 key tiles (2 WMMAs each).
// Pass 2: recompute scores, normalize, write attn probs once to d_out,
//         repack prob tile via 1KB LDS into A layout, 4 WMMAs vs V^T.
// ---------------------------------------------------------------------------
__global__ __launch_bounds__(32) void k_attn(
    const _Float16* __restrict__ Qp, const _Float16* __restrict__ Kp,
    const _Float16* __restrict__ Vtp, float* __restrict__ attn,
    _Float16* __restrict__ CTX) {
  __shared__ _Float16 pbuf[16 * 32];
  const int lane = threadIdx.x, n = lane & 15, hi = lane >> 4;
  const int qb = blockIdx.x * 16, h = blockIdx.y, b = blockIdx.z;
  const size_t head = (size_t)b * NHEAD + h;
  const _Float16* Qh  = Qp  + head * SEQ * DEPTH;
  const _Float16* Kh  = Kp  + head * SEQ * DEPTH;
  const _Float16* Vth = Vtp + head * DEPTH * SEQ;
  float* attn_h = attn + head * (size_t)SEQ * SEQ;
  const float scale = 0.125f;  // 1/sqrt(64)

  // Q tile in A layout (row = lane&15): two 16-byte runs per lane per K-chunk
  U16H aq[2];
  const _Float16* qrow = Qh + (size_t)(qb + (lane & 15)) * DEPTH;
#pragma unroll
  for (int kk = 0; kk < 2; ++kk) {
    aq[kk].h[0] = *(const v8h*)(qrow + kk * 32 + hi * 16);
    aq[kk].h[1] = *(const v8h*)(qrow + kk * 32 + hi * 16 + 8);
  }

  float rmax[8], rsum[8];
#pragma unroll
  for (int r = 0; r < 8; ++r) { rmax[r] = -__builtin_inff(); rsum[r] = 0.f; }

  // ---- Pass 1: online max / sum --------------------------------------
  for (int kt = 0; kt < SEQ / 16; ++kt) {
    const _Float16* kp = Kh + (size_t)(kt * 16 + n) * DEPTH;
    __builtin_prefetch(kp + 16 * DEPTH, 0, 1);  // next key tile -> global_prefetch
    v16h b0 = *(const v16h*)(kp + hi * 16);
    v16h b1 = *(const v16h*)(kp + 32 + hi * 16);
    v8f c = {};
    c = wmma16(aq[0].v, b0, c);
    c = wmma16(aq[1].v, b1, c);
    U8F cu; cu.v = c;
#pragma unroll
    for (int r = 0; r < 8; ++r) {
      float s = cu.e[r] * scale;
      float tm = redmax16(s);
      float mn = fmaxf(rmax[r], tm);
      float corr = __expf(rmax[r] - mn);
      float ts = redsum16(__expf(s - mn));
      rsum[r] = rsum[r] * corr + ts;
      rmax[r] = mn;
    }
  }
  float rinv[8];
#pragma unroll
  for (int r = 0; r < 8; ++r) rinv[r] = 1.0f / rsum[r];

  // ---- Pass 2: normalize, write attn, accumulate ctx = P @ V ---------
  v8f cacc[4] = {};
  for (int kt2 = 0; kt2 < SEQ / 32; ++kt2) {
#pragma unroll
    for (int j = 0; j < 2; ++j) {
      int kt = kt2 * 2 + j;
      const _Float16* kp = Kh + (size_t)(kt * 16 + n) * DEPTH;
      v16h b0 = *(const v16h*)(kp + hi * 16);
      v16h b1 = *(const v16h*)(kp + 32 + hi * 16);
      v8f c = {};
      c = wmma16(aq[0].v, b0, c);
      c = wmma16(aq[1].v, b1, c);
      U8F cu; cu.v = c;
#pragma unroll
      for (int r = 0; r < 8; ++r) {
        int m = hi * 8 + r;
        float p = __expf(cu.e[r] * scale - rmax[r]) * rinv[r];
        attn_h[(size_t)(qb + m) * SEQ + kt * 16 + n] = p;
        pbuf[m * 32 + j * 16 + n] = (_Float16)p;
      }
    }
    __syncthreads();  // single-wave WG: barrier -> S_NOP, orders LDS traffic
    U16H ap;
#pragma unroll
    for (int i = 0; i < 16; ++i)
      ap.e[i] = pbuf[(lane & 15) * 32 + a_kidx(i, hi)];
#pragma unroll
    for (int nt = 0; nt < 4; ++nt) {
      const _Float16* vp = Vth + (size_t)(nt * 16 + n) * SEQ + kt2 * 32;
      v16h bv = *(const v16h*)(vp + hi * 16);
      cacc[nt] = wmma16(ap.v, bv, cacc[nt]);
    }
    __syncthreads();
  }

  // ctx -> CTX[b][s][h*64 + d] (f16, merged heads)
#pragma unroll
  for (int nt = 0; nt < 4; ++nt) {
    int d = nt * 16 + n;
    U8F cu; cu.v = cacc[nt];
#pragma unroll
    for (int r = 0; r < 8; ++r) {
      int row = qb + hi * 8 + r;
      CTX[((size_t)b * SEQ + row) * DMODEL + h * 64 + d] = (_Float16)cu.e[r];
    }
  }
}

// ---------------------------------------------------------------------------
// Kernel 3: out = ctx @ dense_w + dense_b.  grid = 2048 row-tiles, 1 wave.
// K=512 -> 16 chunks x 4 N-tiles = 64 WMMAs per wave.
// ---------------------------------------------------------------------------
__global__ __launch_bounds__(32) void k_out(
    const _Float16* __restrict__ CTX, const _Float16* __restrict__ DT,
    const float* __restrict__ db, float* __restrict__ out) {
  const int lane = threadIdx.x, n = lane & 15, hi = lane >> 4;
  const int rt = blockIdx.x;
  const _Float16* arow = CTX + (size_t)(rt * 16 + (lane & 15)) * DMODEL;

  v8f acc[4] = {};
  for (int kc = 0; kc < DMODEL / 32; ++kc) {
    int kk = kc * 32;
    U16H a;
    a.h[0] = *(const v8h*)(arow + kk + hi * 16);
    a.h[1] = *(const v8h*)(arow + kk + hi * 16 + 8);
#pragma unroll
    for (int nt = 0; nt < 4; ++nt) {
      int col = nt * 16 + n;
      v16h bm = *(const v16h*)(DT + (size_t)col * DMODEL + kk + hi * 16);
      acc[nt] = wmma16(a.v, bm, acc[nt]);
    }
  }
#pragma unroll
  for (int nt = 0; nt < 4; ++nt) {
    int d = nt * 16 + n;
    float bvl = db[d];
    U8F cu; cu.v = acc[nt];
#pragma unroll
    for (int r = 0; r < 8; ++r) {
      int mg = rt * 16 + hi * 8 + r;
      out[(size_t)mg * DEPTH + d] = cu.e[r] + bvl;
    }
  }
}

// ---------------------------------------------------------------------------
extern "C" void kernel_launch(void* const* d_in, const int* in_sizes, int n_in,
                              void* d_out, int out_size, void* d_ws,
                              size_t ws_size, hipStream_t stream) {
  const float* x   = (const float*)d_in[0];
  const float* wq  = (const float*)d_in[1];
  const float* bq  = (const float*)d_in[2];
  const float* wk  = (const float*)d_in[3];
  const float* bk  = (const float*)d_in[4];
  const float* wv  = (const float*)d_in[5];
  const float* bv  = (const float*)d_in[6];
  const float* dw  = (const float*)d_in[7];
  const float* db  = (const float*)d_in[8];
  const float* pos = (const float*)d_in[9];

  float* out  = (float*)d_out;            // 2,097,152 floats
  float* attn = out + OUT_ELEMS;          // 268,435,456 floats

  _Float16* ws  = (_Float16*)d_ws;        // needs ~128.3 MB
  _Float16* Qp  = ws + Q_OFF;
  _Float16* Kp  = ws + K_OFF;
  _Float16* Vtp = ws + VT_OFF;
  _Float16* CTX = ws + CTX_OFF;
  _Float16* WT  = ws + WT_OFF;
  _Float16* DT  = ws + DT_OFF;

  k_prep_w<<<512, 256, 0, stream>>>(wq, wk, wv, dw, WT, DT);
  k_qkv<<<dim3(NROWS / 16, NHEAD, 3), 32, 0, stream>>>(
      x, pos, bq, bk, bv, WT, Qp, Kp, Vtp);
  k_attn<<<dim3(SEQ / 16, NHEAD, NBATCH), 32, 0, stream>>>(
      Qp, Kp, Vtp, attn, CTX);
  k_out<<<NROWS / 16, 32, 0, stream>>>(CTX, DT, db, out);
}
